// GATDenoiser_62380105008143
// MI455X (gfx1250) — matrix operand order repaired and checked by reading, not compiled, not run on hardware
//
#include <hip/hip_runtime.h>
#include <hip/hip_bf16.h>

// ---------------- problem constants ----------------
#define BB   8
#define NN   4096
#define KNN  32
#define BNN  (BB * NN)   // 32768 nodes total

typedef __attribute__((ext_vector_type(16))) _Float16 v16h;
typedef __attribute__((ext_vector_type(8)))  float    v8f;

__device__ __forceinline__ float lrelu(float x, float s) { return x > 0.0f ? x : s * x; }

__device__ __forceinline__ float wave_max(float v) {
    #pragma unroll
    for (int o = 16; o > 0; o >>= 1) v = fmaxf(v, __shfl_xor(v, o, 32));
    return v;
}
__device__ __forceinline__ float wave_sum(float v) {
    #pragma unroll
    for (int o = 16; o > 0; o >>= 1) v += __shfl_xor(v, o, 32);
    return v;
}

// ---- CDNA5 async-copy helpers -------------------------------------------------
__device__ __forceinline__ void async_load_b128_to_lds(unsigned lds_off,
                                                       const void* gaddr) {
    asm volatile("global_load_async_to_lds_b128 %0, %1, off"
                 :
                 : "v"(lds_off), "v"((unsigned long long)(size_t)gaddr)
                 : "memory");
}
template<int N>
__device__ __forceinline__ void wait_asynccnt() {
#if __has_builtin(__builtin_amdgcn_s_wait_asynccnt)
    __builtin_amdgcn_s_wait_asynccnt((short)N);
#else
    if (N == 0)      asm volatile("s_wait_asynccnt 0x0" ::: "memory");
    else if (N == 1) asm volatile("s_wait_asynccnt 0x1" ::: "memory");
    else             asm volatile("s_wait_asynccnt 0x2" ::: "memory");
#endif
}

// ---------------- row |x|^2 ----------------
template<int F>
__global__ void rowsumsq_kernel(const float* __restrict__ x, float* __restrict__ sq) {
    int i = blockIdx.x * blockDim.x + threadIdx.x;
    if (i >= BNN) return;
    float s = 0.0f;
    #pragma unroll
    for (int f = 0; f < F; ++f) { float v = x[(size_t)i * F + f]; s += v * v; }
    sq[i] = s;
}

// ---------------- f32 -> zero-padded f16 panel conversion ----------------
template<int F, int KPAD>
__global__ void cvt_f16_kernel(const float* __restrict__ x, _Float16* __restrict__ xh) {
    int i = blockIdx.x * blockDim.x + threadIdx.x;       // node*KPAD + k
    if (i >= BNN * KPAD) return;
    int node = i / KPAD, k = i % KPAD;
    xh[i] = (k < F) ? (_Float16)x[(size_t)node * F + k] : (_Float16)0.0f;
}

// ---------------- fused Gram-tile (WMMA f16) + top-32 selection ----------------
// 4 waves / block; each wave owns 16 query rows; all waves share a double-buffered
// LDS column panel filled by async global->LDS B128 copies.
template<int KPAD>
__global__ __launch_bounds__(128)
void knn_kernel(const _Float16* __restrict__ xh, const float* __restrict__ sq,
                int* __restrict__ nbr) {
    constexpr int KSTEPS      = KPAD / 32;
    constexpr int PANEL_BYTES = 16 * KPAD * 2;           // 16 nodes, f16
    constexpr int WAITN       = (KPAD == 128) ? 2 : 1;   // async instrs/batch/wave

    const int b    = blockIdx.y;
    const int wave = threadIdx.x >> 5;
    const int lane = threadIdx.x & 31;
    const int tid  = threadIdx.x;
    const int rb   = blockIdx.x * 64 + wave * 16;        // this wave's query rows
    const int nl   = lane & 15;
    const int hi   = lane >> 4;

    const _Float16* xb  = xh + (size_t)b * NN * KPAD;
    const float*    sqb = sq + (size_t)b * NN;

    __shared__ _Float16 panel[2][16 * KPAD];
    __shared__ float    d2s[4][16][17];

    // A fragments (query rows), contiguous k = 16*hi + s layout; packed 32-bit loads.
    union AV { v16h v; unsigned u[8]; } af[KSTEPS];
    #pragma unroll
    for (int kk = 0; kk < KSTEPS; ++kk) {
        #pragma unroll
        for (int v = 0; v < 8; ++v) {
            int k0 = 2 * v + hi * 16 + kk * 32;
            af[kk].u[v] = *(const unsigned*)(xb + (size_t)(rb + nl) * KPAD + k0);
        }
    }

    // hoisted row norms for the 8 output rows this lane produces
    float rs[8];
    #pragma unroll
    for (int v = 0; v < 8; ++v) rs[v] = sqb[rb + v + hi * 8];

    // per-row top-32 (sorted ascending), register resident, owned by lanes 0..15
    float best[KNN]; int bidx[KNN];
    #pragma unroll
    for (int t = 0; t < KNN; ++t) { best[t] = 3.0e38f; bidx[t] = 0; }

    // issue async fill of panel[p] with column tile starting at node cb
    auto issue = [&](int p, int cb) {
        const char* src = (const char*)(xb + (size_t)cb * KPAD);
        unsigned    dst = (unsigned)(size_t)(&panel[p][0]);
        for (int tt = tid; tt * 16 < PANEL_BYTES; tt += 128)
            async_load_b128_to_lds(dst + tt * 16, src + tt * 16);
    };

    issue(0, 0);
    int p = 0;
    for (int cb = 0; cb < NN; cb += 16, p ^= 1) {
        if (cb + 16 < NN) { issue(p ^ 1, cb + 16); wait_asynccnt<WAITN>(); }
        else              { wait_asynccnt<0>(); }
        __syncthreads();                                 // panel[p] ready for all waves

        const _Float16* pan = &panel[p][0];
        v8f acc = {};
        #pragma unroll
        for (int kk = 0; kk < KSTEPS; ++kk) {
            union BV { v16h v; uint4 q[2]; } bf;
            const uint4* ps = (const uint4*)(pan + (size_t)nl * KPAD + kk * 32 + hi * 16);
            bf.q[0] = ps[0];                             // ds_load_b128
            bf.q[1] = ps[1];                             // ds_load_b128
            acc = __builtin_amdgcn_wmma_f32_16x16x32_f16(
                false, af[kk].v, false, bf.v, (short)0, acc, false, false);
        }

        float sqc = sqb[cb + nl];
        #pragma unroll
        for (int v = 0; v < 8; ++v) {
            int m = v + hi * 8;
            float d2 = rs[v] + sqc - 2.0f * acc[v];
            if (rb + m == cb + nl) d2 += 1.0e10f;        // exclude self-loop
            d2s[wave][m][nl] = d2;
        }

        if (lane < 16) {                                 // lane owns query row rb+lane
            #pragma unroll
            for (int c = 0; c < 16; ++c) {
                float val = d2s[wave][lane][c];
                int   idx = cb + c;
                if (val < best[KNN - 1]) {
                    best[KNN - 1] = val; bidx[KNN - 1] = idx;
                    #pragma unroll
                    for (int t = KNN - 1; t > 0; --t) {
                        if (best[t] < best[t - 1]) {
                            float tv = best[t]; best[t] = best[t - 1]; best[t - 1] = tv;
                            int   ti = bidx[t]; bidx[t] = bidx[t - 1]; bidx[t - 1] = ti;
                        }
                    }
                }
            }
        }
        __syncthreads();                                 // all done with panel[p]
    }

    if (lane < 16) {
        int row = rb + lane;
        #pragma unroll
        for (int t = 0; t < KNN; ++t)
            nbr[((size_t)b * NN + row) * KNN + t] = bidx[t];
    }
}

// ---------------- h = x @ W ----------------
template<int FIN, int FOUT>
__global__ void linear_kernel(const float* __restrict__ x, const float* __restrict__ W,
                              float* __restrict__ out) {
    int i = blockIdx.x * blockDim.x + threadIdx.x;
    if (i >= BNN * FOUT) return;
    int node = i / FOUT, o = i % FOUT;
    float s = 0.0f;
    #pragma unroll
    for (int f = 0; f < FIN; ++f) s += x[(size_t)node * FIN + f] * W[f * FOUT + o];
    out[i] = s;
}

// ---------------- attention scores s_src/s_dst ----------------
template<int HEADS, int DOUT>
__global__ void scores_kernel(const float* __restrict__ h, const float* __restrict__ asrc,
                              const float* __restrict__ adst,
                              float* __restrict__ ssrc, float* __restrict__ sdst) {
    int i = blockIdx.x * blockDim.x + threadIdx.x;
    if (i >= BNN * HEADS) return;
    int node = i / HEADS, hh = i % HEADS;
    float s1 = 0.0f, s2 = 0.0f;
    #pragma unroll
    for (int d = 0; d < DOUT; ++d) {
        float v = h[(size_t)node * HEADS * DOUT + hh * DOUT + d];
        s1 += v * asrc[hh * DOUT + d];
        s2 += v * adst[hh * DOUT + d];
    }
    ssrc[i] = s1; sdst[i] = s2;
}

// ---------------- GAT aggregate (concat heads) + bias + activation ----------------
template<int HEADS, int DOUT>
__global__ void gat_agg_concat(const float* __restrict__ h, const float* __restrict__ ssrc,
                               const float* __restrict__ sdst, const int* __restrict__ nbr,
                               const float* __restrict__ bias, float slope, float act_slope,
                               float* __restrict__ out) {
    const int HD   = HEADS * DOUT;
    const int node = blockIdx.x * (blockDim.x >> 5) + (threadIdx.x >> 5);
    const int lane = threadIdx.x & 31;
    const int b    = node / NN;
    int j  = nbr[(size_t)node * KNN + lane];
    int jg = b * NN + j;                       // global neighbor node id

    #pragma unroll
    for (int hh = 0; hh < HEADS; ++hh) {
        float e = lrelu(ssrc[(size_t)jg * HEADS + hh] + sdst[(size_t)node * HEADS + hh], slope);
        float m = wave_max(e);
        float p = __expf(e - m);
        float s = wave_sum(p);
        float alpha = p / s;
        for (int d0 = 0; d0 < DOUT; d0 += 32) {
            int d = d0 + lane;
            float acc = 0.0f;
            #pragma unroll
            for (int k = 0; k < 32; ++k) {
                float a  = __shfl(alpha, k, 32);
                int   jk = __shfl(jg,    k, 32);
                if (d < DOUT) acc += a * h[(size_t)jk * HD + hh * DOUT + d];
            }
            if (d < DOUT) {
                int c = hh * DOUT + d;
                out[(size_t)node * HD + c] = lrelu(acc + bias[c], act_slope);
            }
        }
    }
}

// ---------------- GAT aggregate (mean over heads) + bias ----------------
template<int HEADS, int DOUT>
__global__ void gat_agg_mean(const float* __restrict__ h, const float* __restrict__ ssrc,
                             const float* __restrict__ sdst, const int* __restrict__ nbr,
                             const float* __restrict__ bias, float slope,
                             float* __restrict__ out) {
    const int HD   = HEADS * DOUT;
    const int node = blockIdx.x * (blockDim.x >> 5) + (threadIdx.x >> 5);
    const int lane = threadIdx.x & 31;
    const int b    = node / NN;
    int j  = nbr[(size_t)node * KNN + lane];
    int jg = b * NN + j;
    int d  = lane;                              // DOUT <= 32
    float accsum = 0.0f;
    #pragma unroll
    for (int hh = 0; hh < HEADS; ++hh) {
        float e = lrelu(ssrc[(size_t)jg * HEADS + hh] + sdst[(size_t)node * HEADS + hh], slope);
        float m = wave_max(e);
        float p = __expf(e - m);
        float s = wave_sum(p);
        float alpha = p / s;
        float acc = 0.0f;
        #pragma unroll
        for (int k = 0; k < 32; ++k) {
            float a  = __shfl(alpha, k, 32);
            int   jk = __shfl(jg,    k, 32);
            if (d < DOUT) acc += a * h[(size_t)jk * HD + hh * DOUT + d];
        }
        accsum += acc;
    }
    if (d < DOUT)
        out[(size_t)node * DOUT + d] = accsum / (float)HEADS + bias[d];
}

// ---------------- BatchNorm (training mode, biased var): stats then apply ----------------
template<int C>
__global__ void bn_stats_kernel(const float* __restrict__ x,
                                float* __restrict__ mean, float* __restrict__ var) {
    __shared__ float s1[256], s2[256];
    int c = blockIdx.x, t = threadIdx.x;
    float a = 0.0f, q = 0.0f;
    for (int r = t; r < BNN; r += 256) {
        float v = x[(size_t)r * C + c];
        a += v; q += v * v;
    }
    s1[t] = a; s2[t] = q; __syncthreads();
    for (int o = 128; o > 0; o >>= 1) {
        if (t < o) { s1[t] += s1[t + o]; s2[t] += s2[t + o]; }
        __syncthreads();
    }
    if (t == 0) {
        float m = s1[0] / (float)BNN;
        mean[c] = m;
        var[c]  = s2[0] / (float)BNN - m * m;
    }
}

template<int C>
__global__ void bn_apply_kernel(const float* __restrict__ x, const float* __restrict__ mean,
                                const float* __restrict__ var, const float* __restrict__ g,
                                const float* __restrict__ be, float* __restrict__ y) {
    int i = blockIdx.x * blockDim.x + threadIdx.x;
    if (i >= BNN * C) return;
    int c = i % C;
    y[i] = (x[i] - mean[c]) * rsqrtf(var[c] + 1e-5f) * g[c] + be[c];
}

// ---------------- loss ----------------
__global__ void zero_kernel(float* p) { *p = 0.0f; }

__global__ void loss_kernel(const float* __restrict__ h, const float* __restrict__ y,
                            float* __restrict__ loss) {
    __shared__ float s[256];
    const int total = BNN * 6;
    int i = blockIdx.x * 256 + threadIdx.x;
    float v = 0.0f;
    if (i < total) { float d = h[i] - y[i]; v = d * d / (float)total; }
    s[threadIdx.x] = v; __syncthreads();
    for (int o = 128; o > 0; o >>= 1) {
        if (threadIdx.x < o) s[threadIdx.x] += s[threadIdx.x + o];
        __syncthreads();
    }
    if (threadIdx.x == 0) atomicAdd(loss, s[0]);
}

// ---------------- launch ----------------
extern "C" void kernel_launch(void* const* d_in, const int* in_sizes, int n_in,
                              void* d_out, int out_size, void* d_ws, size_t ws_size,
                              hipStream_t stream) {
    const float* x      = (const float*)d_in[0];
    const float* y      = (const float*)d_in[1];
    const float* W0     = (const float*)d_in[2];
    const float* a_src0 = (const float*)d_in[3];
    const float* a_dst0 = (const float*)d_in[4];
    const float* b0     = (const float*)d_in[5];
    const float* g0     = (const float*)d_in[6];
    const float* be0    = (const float*)d_in[7];
    const float* W1     = (const float*)d_in[8];
    const float* a_src1 = (const float*)d_in[9];
    const float* a_dst1 = (const float*)d_in[10];
    const float* b1     = (const float*)d_in[11];
    const float* g1     = (const float*)d_in[12];
    const float* be1    = (const float*)d_in[13];
    const float* W2     = (const float*)d_in[14];
    const float* a_src2 = (const float*)d_in[15];
    const float* a_dst2 = (const float*)d_in[16];
    const float* b2     = (const float*)d_in[17];
    float* out = (float*)d_out;

    // workspace carve-up
    char* w = (char*)d_ws;
    float*     sq    = (float*)w;             w += (size_t)BNN * 4;
    int*       nbr   = (int*)w;               w += (size_t)BNN * KNN * 4;
    _Float16*  xhp   = (_Float16*)w;          w += (size_t)BNN * 128 * 2;   // padded f16 panels
    float*     bufH  = (float*)w;             w += (size_t)BNN * 128 * 4;
    float*     bufA  = (float*)w;             w += (size_t)BNN * 128 * 4;   // pre-BN activations
    float*     bufX  = (float*)w;             w += (size_t)BNN * 128 * 4;   // layer inputs
    float*     ssrc  = (float*)w;             w += (size_t)BNN * 8 * 4;
    float*     sdst  = (float*)w;             w += (size_t)BNN * 8 * 4;
    float*     meanv = (float*)w;             w += 128 * 4;
    float*     varv  = (float*)w;             w += 128 * 4;

    const dim3 knn_grid(NN / 64, BB);         // 4 waves/block, 16 rows/wave
    const int  agg_blocks = BNN / 8;          // 8 waves (nodes) per 256-thread block

    // ================= layer 0: 6 -> 4 heads x 16 (concat) =================
    rowsumsq_kernel<6><<<(BNN + 255) / 256, 256, 0, stream>>>(x, sq);
    cvt_f16_kernel<6, 32><<<(BNN * 32 + 255) / 256, 256, 0, stream>>>(x, xhp);
    knn_kernel<32><<<knn_grid, 128, 0, stream>>>(xhp, sq, nbr);
    linear_kernel<6, 64><<<(BNN * 64 + 255) / 256, 256, 0, stream>>>(x, W0, bufH);
    scores_kernel<4, 16><<<(BNN * 4 + 255) / 256, 256, 0, stream>>>(bufH, a_src0, a_dst0, ssrc, sdst);
    gat_agg_concat<4, 16><<<agg_blocks, 256, 0, stream>>>(bufH, ssrc, sdst, nbr, b0, 0.2f, 0.2f, bufA);
    bn_stats_kernel<64><<<64, 256, 0, stream>>>(bufA, meanv, varv);
    bn_apply_kernel<64><<<(BNN * 64 + 255) / 256, 256, 0, stream>>>(bufA, meanv, varv, g0, be0, bufX);

    // ================= layer 1: 64 -> 2 heads x 64 (concat) =================
    rowsumsq_kernel<64><<<(BNN + 255) / 256, 256, 0, stream>>>(bufX, sq);
    cvt_f16_kernel<64, 64><<<(BNN * 64 + 255) / 256, 256, 0, stream>>>(bufX, xhp);
    knn_kernel<64><<<knn_grid, 128, 0, stream>>>(xhp, sq, nbr);
    linear_kernel<64, 128><<<(BNN * 128 + 255) / 256, 256, 0, stream>>>(bufX, W1, bufH);
    scores_kernel<2, 64><<<(BNN * 2 + 255) / 256, 256, 0, stream>>>(bufH, a_src1, a_dst1, ssrc, sdst);
    gat_agg_concat<2, 64><<<agg_blocks, 256, 0, stream>>>(bufH, ssrc, sdst, nbr, b1, 0.2f, 0.2f, bufA);
    bn_stats_kernel<128><<<128, 256, 0, stream>>>(bufA, meanv, varv);
    bn_apply_kernel<128><<<(BNN * 128 + 255) / 256, 256, 0, stream>>>(bufA, meanv, varv, g1, be1, bufX);

    // ================= layer 2: 128 -> 8 heads x 6 (mean), slope=1.0 =================
    rowsumsq_kernel<128><<<(BNN + 255) / 256, 256, 0, stream>>>(bufX, sq);
    cvt_f16_kernel<128, 128><<<(BNN * 128 + 255) / 256, 256, 0, stream>>>(bufX, xhp);
    knn_kernel<128><<<knn_grid, 128, 0, stream>>>(xhp, sq, nbr);
    linear_kernel<128, 48><<<(BNN * 48 + 255) / 256, 256, 0, stream>>>(bufX, W2, bufH);
    scores_kernel<8, 6><<<(BNN * 8 + 255) / 256, 256, 0, stream>>>(bufH, a_src2, a_dst2, ssrc, sdst);
    gat_agg_mean<8, 6><<<agg_blocks, 256, 0, stream>>>(bufH, ssrc, sdst, nbr, b2, 1.0f, out);

    // ================= loss =================
    zero_kernel<<<1, 1, 0, stream>>>(out + (size_t)BNN * 6);
    loss_kernel<<<(BNN * 6 + 255) / 256, 256, 0, stream>>>(out, y, out + (size_t)BNN * 6);
}
